// MeanField_49821620634291
// MI455X (gfx1250) — 8-bit matrix operand order repaired and checked by reading, compiled, not run on hardware
//
#include <hip/hip_runtime.h>
#include <math.h>

#define BATCH 16
#define IMG_H 256
#define IMG_W 256
#define HW    (IMG_H * IMG_W)
#define NPIX  (BATCH * HW)

#define TS 32
#define TW 36
#define TE (TW * TW)
#define SKEW 2
#define TP (TE + 8)

typedef int v8i __attribute__((ext_vector_type(8)));

__device__ __forceinline__ unsigned lds_off(const void* p) {
    return (unsigned)(uintptr_t)p;
}
__device__ __forceinline__ void async_load_b32(unsigned lds, const void* gptr) {
    asm volatile("global_load_async_to_lds_b32 %0, %1, off"
                 :: "v"(lds), "v"((unsigned long long)(uintptr_t)gptr) : "memory");
}
__device__ __forceinline__ void async_load_b64(unsigned lds, const void* gptr) {
    asm volatile("global_load_async_to_lds_b64 %0, %1, off"
                 :: "v"(lds), "v"((unsigned long long)(uintptr_t)gptr) : "memory");
}
__device__ __forceinline__ void async_load_b128(unsigned lds, const void* gptr) {
    asm volatile("global_load_async_to_lds_b128 %0, %1, off"
                 :: "v"(lds), "v"((unsigned long long)(uintptr_t)gptr) : "memory");
}
__device__ __forceinline__ void wait_asynccnt0() {
    asm volatile("s_wait_asynccnt 0" ::: "memory");
}

__global__ __launch_bounds__(256) void init_kernel(
    const float* __restrict__ seg, float* __restrict__ xa, float* __restrict__ xb)
{
    int i = blockIdx.x * blockDim.x + threadIdx.x;
    if (i < NPIX) {
        float v = fminf(fmaxf(seg[i], 0.1f), 0.9f);
        xa[i] = v;
        xb[i] = v;
    }
}

__global__ __launch_bounds__(256) void crf_iter_kernel(
    const float* __restrict__ fm,
    const float* __restrict__ dep,
    const float* __restrict__ xa_in,
    const float* __restrict__ xb_in,
    float* __restrict__ xa_out,
    float* __restrict__ xb_out)
{
    __shared__ __align__(16) float sR[TP], sG[TP], sB[TP], sD[TP];
    __shared__ __align__(16) float la0[TP], la1[TP], lb0[TP], lb1[TP];

    const int tid = threadIdx.x;
    const int tilesPerRow = IMG_W / TS;
    const int tilesPerImg = (IMG_H / TS) * tilesPerRow;
    const int img = blockIdx.x / tilesPerImg;
    const int t   = blockIdx.x % tilesPerImg;
    const int ty0 = (t / tilesPerRow) * TS;
    const int tx0 = (t % tilesPerRow) * TS;
    const bool xint = (tx0 != 0) && (tx0 != IMG_W - TS);

    const float* gpl[6] = {
        fm + ((size_t)img * 3 + 0) * HW,
        fm + ((size_t)img * 3 + 1) * HW,
        fm + ((size_t)img * 3 + 2) * HW,
        dep   + (size_t)img * HW,
        xa_in + (size_t)img * HW,
        xb_in + (size_t)img * HW };
    float* spl[6] = { sR, sG, sB, sD, la1, lb1 };

    for (int task = tid; task < 6 * TW; task += 256) {
        const int p  = task / TW;
        const int ly = task % TW;
        const int gy = ty0 + ly - 2;
        float* sp = spl[p];
        const float sval = (p == 0) ? -1.0e4f : 0.0f;
        const int rowbase = ly * TW + SKEW;

        if ((unsigned)gy >= IMG_H) {
            for (int lx = 0; lx < TW; ++lx) sp[rowbase + lx] = sval;
        } else if (xint) {
            const float* gr = gpl[p] + (size_t)gy * IMG_W + (tx0 - 2);
            async_load_b64(lds_off(&sp[rowbase + 0]), gr + 0);
#pragma unroll
            for (int lx = 2; lx < 34; lx += 4)
                async_load_b128(lds_off(&sp[rowbase + lx]), gr + lx);
            async_load_b64(lds_off(&sp[rowbase + 34]), gr + 34);
        } else {
            const float* gr = gpl[p] + (size_t)gy * IMG_W;
            for (int lx = 0; lx < TW; ++lx) {
                const int gx = tx0 + lx - 2;
                if ((unsigned)gx < IMG_W)
                    async_load_b32(lds_off(&sp[rowbase + lx]), gr + gx);
                else
                    sp[rowbase + lx] = sval;
            }
        }
    }
    wait_asynccnt0();
    __syncthreads();

    for (int i = tid; i < TE; i += 256) {
        const int ly = i / TW, lx = i % TW;
        const int idx = i + SKEW;
        const int gy = ty0 + ly - 2, gx = tx0 + lx - 2;
        if ((unsigned)gy < IMG_H && (unsigned)gx < IMG_W) {
            const float xav = la1[idx], xbv = lb1[idx];
            la0[idx] = -__logf(1.0f - xav);
            la1[idx] = -__logf(xav);
            lb0[idx] = -__logf(1.0f - xbv);
            lb1[idx] = -__logf(xbv);
        } else {
            la0[idx] = 0.0f; la1[idx] = 0.0f; lb0[idx] = 0.0f; lb1[idx] = 0.0f;
        }
    }
    __syncthreads();

    const int py  = tid >> 3;
    const int px0 = (tid & 7) << 2;
    float oa[4], ob[4];
#pragma unroll
    for (int q = 0; q < 4; ++q) {
        const int px = px0 + q;
        const int c  = (py + 2) * TW + (px + 2) + SKEW;
        const float cR = sR[c], cG = sG[c], cB = sB[c], cD = sD[c];
        float sa0 = 0.f, sa1 = 0.f, sb0 = 0.f, sb1 = 0.f;
#pragma unroll
        for (int dy = 0; dy < 5; ++dy) {
#pragma unroll
            for (int dx = 0; dx < 5; ++dx) {
                const int n = (py + dy) * TW + (px + dx) + SKEW;
                const float dr  = sR[n] - cR;
                const float dg  = sG[n] - cG;
                const float dbv = sB[n] - cB;
                const float w   = __expf(-50.0f * (dr * dr + dg * dg + dbv * dbv));
                const float dd  = sD[n] - cD;
                const float wd  = w * __expf(-50.0f * dd * dd);
                sa0 += w  * la0[n];  sa1 += w  * la1[n];
                sb0 += wd * lb0[n];  sb1 += wd * lb1[n];
            }
        }
        const float ea0 = __expf(-sa0), ea1 = __expf(-sa1);
        const float eb0 = __expf(-sb0), eb1 = __expf(-sb1);
        float va = ea1 / (1e-6f + ea0 + ea1);
        float vb = eb1 / (1e-6f + eb0 + eb1);
        oa[q] = fminf(fmaxf(va, 0.1f), 0.9f);
        ob[q] = fminf(fmaxf(vb, 0.1f), 0.9f);
    }
    const size_t g0 = (size_t)img * HW + (size_t)(ty0 + py) * IMG_W + (tx0 + px0);
    *(float4*)(xa_out + g0) = make_float4(oa[0], oa[1], oa[2], oa[3]);
    *(float4*)(xb_out + g0) = make_float4(ob[0], ob[1], ob[2], ob[3]);
}

__global__ __launch_bounds__(256) void mask_prep_kernel(
    const float* __restrict__ seg,
    const float* __restrict__ targets,
    const float* __restrict__ sam,
    const float* __restrict__ xa,
    const float* __restrict__ xb,
    float* __restrict__ out_mask,
    unsigned char* __restrict__ mb)
{
    int i = blockIdx.x * blockDim.x + threadIdx.x;
    if (i < NPIX) {
        const unsigned char t  = targets[i] != 0.0f;
        const unsigned char m0 = seg[i] > 0.5f;
        const unsigned char m1 = xa[i] > 0.5f;
        const unsigned char m2 = xb[i] > 0.5f;
        const unsigned char m3 = sam[i] != 0.0f;
        out_mask[i] = m1 ? 1.0f : 0.0f;
        mb[0 * (size_t)NPIX + i] = t;
        mb[1 * (size_t)NPIX + i] = m0;
        mb[2 * (size_t)NPIX + i] = m1;
        mb[3 * (size_t)NPIX + i] = m2;
        mb[4 * (size_t)NPIX + i] = m3;
    }
}

__global__ __launch_bounds__(256) void iou_wmma_kernel(
    const unsigned char* __restrict__ mb,
    float* __restrict__ iou)
{
    __shared__ int sh[9];
    const int img  = blockIdx.x;
    const int tid  = threadIdx.x;
    const int lane = tid & 31;
    const int wave = tid >> 5;
    if (tid < 9) sh[tid] = 0;
    __syncthreads();

    const int r        = lane & 15;
    const int lanehalf = lane >> 4;

    v8i acc = {};
    const size_t wbase = (size_t)img * HW + (size_t)wave * (HW / 8);

    for (int ch = 0; ch < (HW / 8) / 64; ++ch) {
        const size_t cb = wbase + (size_t)ch * 64;

        v8i a = {};
        if (r < 5) {
            const unsigned char* ap = mb + (size_t)r * NPIX + cb + lanehalf * 16;
            const uint4 p0 = *(const uint4*)(ap +  0);
            const uint4 p1 = *(const uint4*)(ap + 32);
            a[0] = (int)p0.x; a[1] = (int)p0.y; a[2] = (int)p0.z; a[3] = (int)p0.w;
            a[4] = (int)p1.x; a[5] = (int)p1.y; a[6] = (int)p1.z; a[7] = (int)p1.w;
        }
        v8i bm = {};
        if (r == 0) {
            const unsigned char* bp = mb + cb + lanehalf * 16;
            const uint4 q0 = *(const uint4*)(bp +  0);
            const uint4 q1 = *(const uint4*)(bp + 32);
            bm[0] = (int)q0.x; bm[1] = (int)q0.y; bm[2] = (int)q0.z; bm[3] = (int)q0.w;
            bm[4] = (int)q1.x; bm[5] = (int)q1.y; bm[6] = (int)q1.z; bm[7] = (int)q1.w;
        } else if (r == 1) {
            bm[0] = bm[1] = bm[2] = bm[3] = 0x01010101;
            bm[4] = bm[5] = bm[6] = bm[7] = 0x01010101;
        }
        acc = __builtin_amdgcn_wmma_i32_16x16x64_iu8(false, a, false, bm, acc,
                                                     false, false);
    }

    if (lane == 0) {
        atomicAdd(&sh[5], acc[1]); atomicAdd(&sh[6], acc[2]);
        atomicAdd(&sh[7], acc[3]); atomicAdd(&sh[8], acc[4]);
    } else if (lane == 1) {
        atomicAdd(&sh[0], acc[0]);
        atomicAdd(&sh[1], acc[1]); atomicAdd(&sh[2], acc[2]);
        atomicAdd(&sh[3], acc[3]); atomicAdd(&sh[4], acc[4]);
    }
    __syncthreads();

    if (tid < 4) {
        const float inter = (float)sh[5 + tid];
        const float uni   = (float)(sh[0] + sh[1 + tid] - sh[5 + tid]);
        iou[img * 4 + tid] = inter / (uni + 1e-6f);
    }
}

__global__ void mean_kernel(const float* __restrict__ iou, float* __restrict__ out4)
{
    const int k = threadIdx.x;
    if (k < 4) {
        float s = 0.0f;
        for (int b = 0; b < BATCH; ++b) s += iou[b * 4 + k];
        out4[k] = s * (1.0f / BATCH);
    }
}

extern "C" void kernel_launch(void* const* d_in, const int* in_sizes, int n_in,
                              void* d_out, int out_size, void* d_ws, size_t ws_size,
                              hipStream_t stream)
{
    (void)in_sizes; (void)n_in; (void)out_size; (void)ws_size;
    const float* fm  = (const float*)d_in[0];
    const float* seg = (const float*)d_in[1];
    const float* dep = (const float*)d_in[2];
    const float* tgt = (const float*)d_in[3];
    const float* sam = (const float*)d_in[4];
    float* out = (float*)d_out;

    float* ws = (float*)d_ws;
    float* xa[2] = { ws,                      ws + (size_t)NPIX };
    float* xb[2] = { ws + 2 * (size_t)NPIX,   ws + 3 * (size_t)NPIX };
    float* iou   = ws + 4 * (size_t)NPIX;
    unsigned char* mb = (unsigned char*)(ws + 4 * (size_t)NPIX + 64);

    init_kernel<<<(NPIX + 255) / 256, 256, 0, stream>>>(seg, xa[0], xb[0]);

    const int nblocks = BATCH * (IMG_H / TS) * (IMG_W / TS);
    for (int it = 0; it < 10; ++it) {
        const int src = it & 1, dst = src ^ 1;
        crf_iter_kernel<<<nblocks, 256, 0, stream>>>(
            fm, dep, xa[src], xb[src], xa[dst], xb[dst]);
    }
    mask_prep_kernel<<<(NPIX + 255) / 256, 256, 0, stream>>>(
        seg, tgt, sam, xa[0], xb[0], out, mb);
    iou_wmma_kernel<<<BATCH, 256, 0, stream>>>(mb, iou);
    mean_kernel<<<1, 32, 0, stream>>>(iou, out + (size_t)NPIX);
}
